// ComicAttentionBlock_47811575939254
// MI455X (gfx1250) — matrix operand order repaired and checked by reading, not compiled, not run on hardware
//
#include <hip/hip_runtime.h>
#include <hip/hip_bf16.h>

typedef __attribute__((ext_vector_type(16))) __bf16 v16bf;
typedef __attribute__((ext_vector_type(8)))  __bf16 v8bf;
typedef __attribute__((ext_vector_type(8)))  float  v8f;
typedef __attribute__((ext_vector_type(4)))  unsigned int u32x4;
typedef __attribute__((ext_vector_type(8)))  int i32x8;
typedef __attribute__((ext_vector_type(4)))  int i32x4;

#define B_  4
#define N_  4096
#define C_  256
#define CD_ 32
#define NCH (N_ / 32)   // 128 key chunks of 32
#define VROWB 80        // padded LDS row stride: 64B data + 16B TDM pad
#define VBUFB (C_ * VROWB)

// K-index of element pair base for 16-bit A operand (16x32), contiguous order:
// VGPR v in 0..7: K = 16*hi + 2v
__device__ __forceinline__ int a_kidx(int v, int hi) {
  return 16 * hi + 2 * v;
}

// ---------------------------------------------------------------------------
// TDM: load a 32(keys) x 256(channels) bf16 tile of hhT[b] into LDS, with
// 16B padding after each 64B row (LDS row stride 80B). Descriptor per ISA §8.
// ---------------------------------------------------------------------------
__device__ __forceinline__ void tdm_issue_v(const __bf16* tile_start,
                                            unsigned int lds_addr) {
  const unsigned long long ga = (unsigned long long)(size_t)tile_start;
  const unsigned long long g0lo =
      1ull /*count=1, is_restore=0, gather=0*/ |
      ((unsigned long long)lds_addr << 32);
  const unsigned long long g0hi =
      (ga & ((1ull << 57) - 1)) | (2ull << 62) /*type=2 image*/;
  u32x4 g0;
  g0[0] = (unsigned)g0lo;  g0[1] = (unsigned)(g0lo >> 32);
  g0[2] = (unsigned)g0hi;  g0[3] = (unsigned)(g0hi >> 32);

  i32x8 g1;
  // data_size=1 (2B), pad_enable, pad_interval code 3 (16 dwords = 64B),
  // pad_amount code 3 (4 dwords = 16B)
  g1[0] = (1 << 16) | (1 << 20) | (3 << 22) | (3 << 25);
  g1[1] = (int)((N_ & 0xffff) << 16);                    // tensor_dim0 lo16
  g1[2] = (int)((N_ >> 16) | ((C_ & 0xffff) << 16));     // dim0 hi | dim1 lo
  g1[3] = (int)((C_ >> 16) | (32u << 16));               // dim1 hi | tile_dim0=32
  g1[4] = C_;                                            // tile_dim1=256, tile_dim2=0
  g1[5] = N_;                                            // tensor_dim0_stride lo32
  g1[6] = 0;                                             // stride hi | dim1_stride lo16
  g1[7] = (int)((unsigned)(N_ * C_) >> 16);              // tensor_dim1_stride[47:16]
  i32x4 z4; z4[0] = 0; z4[1] = 0; z4[2] = 0; z4[3] = 0;
#if defined(__clang_major__) && __clang_major__ >= 23
  i32x8 z8;
  #pragma unroll
  for (int i = 0; i < 8; ++i) z8[i] = 0;
  __builtin_amdgcn_tensor_load_to_lds(g0, g1, z4, z4, z8, 0);
#else
  __builtin_amdgcn_tensor_load_to_lds(g0, g1, z4, z4, 0);
#endif
}

// ---------------------------------------------------------------------------
// Projection f/g: Y[16384 x 32] = X[16384 x 256] @ W[256 x 32] + b, bf16 out.
// grid = (1024, 2, 2), block = 32
// ---------------------------------------------------------------------------
__global__ void proj_fg_kernel(const float* __restrict__ x,
                               const float* __restrict__ Wf, const float* __restrict__ bfp,
                               const float* __restrict__ Wg, const float* __restrict__ bgp,
                               __bf16* __restrict__ fbuf, __bf16* __restrict__ gbuf) {
  const int lane = threadIdx.x & 31;
  const int m    = lane & 15;
  const int hi   = lane >> 4;
  const int R    = blockIdx.x * 16;
  const int Cb   = blockIdx.y * 16;
  const float* W    = (blockIdx.z == 0) ? Wf  : Wg;
  const float* bias = (blockIdx.z == 0) ? bfp : bgp;
  __bf16*      outp = (blockIdx.z == 0) ? fbuf : gbuf;

  v8f acc;
  #pragma unroll
  for (int r = 0; r < 8; ++r) acc[r] = 0.0f;

  for (int kk = 0; kk < C_; kk += 32) {
    v16bf a;
    const float* xrow = x + (size_t)(R + m) * C_ + kk;
    #pragma unroll
    for (int v = 0; v < 8; ++v) {
      const int K = a_kidx(v, hi);
      const float2 t = *(const float2*)(xrow + K);
      a[2 * v]     = (__bf16)t.x;
      a[2 * v + 1] = (__bf16)t.y;
    }
    v16bf bb;
    #pragma unroll
    for (int i = 0; i < 16; ++i)
      bb[i] = (__bf16)W[(size_t)(kk + i + 16 * hi) * CD_ + Cb + m];
    acc = __builtin_amdgcn_wmma_f32_16x16x32_bf16(false, a, false, bb, (short)0,
                                                  acc, false, false);
  }
  const float bv = bias[Cb + m];
  #pragma unroll
  for (int r = 0; r < 8; ++r) {
    const int row = R + r + 8 * hi;
    outp[(size_t)row * CD_ + Cb + m] = (__bf16)(acc[r] + bv);
  }
}

// ---------------------------------------------------------------------------
// Projection hh stored TRANSPOSED: hhT[b][c][n], bf16. grid = (1024,16), blk 32
// ---------------------------------------------------------------------------
__global__ void proj_h_kernel(const float* __restrict__ x,
                              const float* __restrict__ Wh, const float* __restrict__ bhp,
                              __bf16* __restrict__ hhT) {
  const int lane = threadIdx.x & 31;
  const int m    = lane & 15;
  const int hi   = lane >> 4;
  const int R    = blockIdx.x * 16;
  const int Cb   = blockIdx.y * 16;

  v8f acc;
  #pragma unroll
  for (int r = 0; r < 8; ++r) acc[r] = 0.0f;

  for (int kk = 0; kk < C_; kk += 32) {
    v16bf a;
    const float* xrow = x + (size_t)(R + m) * C_ + kk;
    #pragma unroll
    for (int v = 0; v < 8; ++v) {
      const int K = a_kidx(v, hi);
      const float2 t = *(const float2*)(xrow + K);
      a[2 * v]     = (__bf16)t.x;
      a[2 * v + 1] = (__bf16)t.y;
    }
    v16bf bb;
    #pragma unroll
    for (int i = 0; i < 16; ++i)
      bb[i] = (__bf16)Wh[(size_t)(kk + i + 16 * hi) * C_ + Cb + m];
    acc = __builtin_amdgcn_wmma_f32_16x16x32_bf16(false, a, false, bb, (short)0,
                                                  acc, false, false);
  }
  const int c  = Cb + m;
  const int b  = R / N_;
  const int nl = (R % N_) + 8 * hi;
  const float bv = bhp[c];
  v8bf o;
  #pragma unroll
  for (int r = 0; r < 8; ++r) o[r] = (__bf16)(acc[r] + bv);
  *(v8bf*)(hhT + ((size_t)b * C_ + c) * N_ + nl) = o;
}

// ---------------------------------------------------------------------------
// Flash attention. 4 waves/block, one 16-row query tile each, same batch,
// lock-step over 128 chunks of 32 keys. V chunks staged to LDS by the TDM
// (double buffered, TENSORcnt-synced); S = Q@K^T and O += P@V via bf16 WMMA.
// grid = 256, block = 128
// ---------------------------------------------------------------------------
__global__ void __launch_bounds__(128, 1)
attn_kernel(const float* __restrict__ x,
            const __bf16* __restrict__ fbuf, const __bf16* __restrict__ gbuf,
            const __bf16* __restrict__ hhT, float* __restrict__ outp) {
  const int lane = threadIdx.x & 31;
  const int wave = threadIdx.x >> 5;
  const int tile = blockIdx.x * 4 + wave;   // 1024 query tiles
  const int b    = tile >> 8;
  const int q0   = (tile & 255) * 16;
  const int m    = lane & 15;
  const int hi   = lane >> 4;

  __shared__ __align__(32) unsigned char vstage[2][VBUFB]; // 2 x 20 KB V chunks
  __shared__ __bf16 plds[4][16][32];                       // per-wave P staging

  const __bf16* fb = fbuf + (size_t)b * N_ * CD_;
  const __bf16* hb = hhT  + (size_t)b * C_ * N_;

  // Q (from g) in A layout
  v16bf qa;
  const __bf16* grow = gbuf + (size_t)(b * N_ + q0 + m) * CD_;
  #pragma unroll
  for (int v = 0; v < 8; ++v) {
    const int K = a_kidx(v, hi);
    qa[2 * v]     = grow[K];
    qa[2 * v + 1] = grow[K + 1];
  }

  v8f acc[16];
  #pragma unroll
  for (int t = 0; t < 16; ++t)
    #pragma unroll
    for (int r = 0; r < 8; ++r) acc[t][r] = 0.0f;

  float mrun[8], lrun[8];
  #pragma unroll
  for (int r = 0; r < 8; ++r) { mrun[r] = -3.0e38f; lrun[r] = 0.0f; }

  const unsigned int lds_v0 = (unsigned int)(size_t)&vstage[0][0];
  const unsigned int lds_v1 = (unsigned int)(size_t)&vstage[1][0];

  // prologue: stage chunk 0
  if (wave == 0) tdm_issue_v(hb, lds_v0);

  for (int j = 0; j < NCH; ++j) {
    const int k0 = j * 32;
    if (wave == 0) {
      if (j + 1 < NCH) {       // stage next chunk into the other buffer,
        tdm_issue_v(hb + (size_t)(j + 1) * 32, ((j + 1) & 1) ? lds_v1 : lds_v0);
        __builtin_amdgcn_s_wait_tensorcnt(1);   // chunk j has landed
      } else {
        __builtin_amdgcn_s_wait_tensorcnt(0);
      }
    }
    __syncthreads();           // publish chunk j to all waves

    if (j + 1 < NCH)
      __builtin_prefetch(fb + (size_t)(k0 + 32 + m) * CD_, 0, 1);

    // S = Q @ K^T  (f rows are the B operand; contiguous 32B per lane)
    const v16bf kb0 = *(const v16bf*)(fb + (size_t)(k0 + m) * CD_ + 16 * hi);
    const v16bf kb1 = *(const v16bf*)(fb + (size_t)(k0 + 16 + m) * CD_ + 16 * hi);
    v8f z;
    #pragma unroll
    for (int r = 0; r < 8; ++r) z[r] = 0.0f;
    v8f sa = __builtin_amdgcn_wmma_f32_16x16x32_bf16(false, qa, false, kb0,
                                                     (short)0, z, false, false);
    v8f sb = __builtin_amdgcn_wmma_f32_16x16x32_bf16(false, qa, false, kb1,
                                                     (short)0, z, false, false);

    // online softmax (row r+8*hi lives in one 16-lane half; xor bits 0..3)
    float p0[8], p1[8], corr[8];
    #pragma unroll
    for (int r = 0; r < 8; ++r) {
      float cm = fmaxf(sa[r], sb[r]);
      cm = fmaxf(cm, __shfl_xor(cm, 1, 32));
      cm = fmaxf(cm, __shfl_xor(cm, 2, 32));
      cm = fmaxf(cm, __shfl_xor(cm, 4, 32));
      cm = fmaxf(cm, __shfl_xor(cm, 8, 32));
      const float mn = fmaxf(mrun[r], cm);
      corr[r] = __expf(mrun[r] - mn);
      p0[r]   = __expf(sa[r] - mn);
      p1[r]   = __expf(sb[r] - mn);
      float rs = p0[r] + p1[r];
      rs += __shfl_xor(rs, 1, 32);
      rs += __shfl_xor(rs, 2, 32);
      rs += __shfl_xor(rs, 4, 32);
      rs += __shfl_xor(rs, 8, 32);
      lrun[r] = lrun[r] * corr[r] + rs;
      mrun[r] = mn;
    }
    #pragma unroll
    for (int t = 0; t < 16; ++t)
      #pragma unroll
      for (int r = 0; r < 8; ++r) acc[t][r] *= corr[r];

    // P: C-layout -> bf16 -> LDS -> A-layout
    #pragma unroll
    for (int r = 0; r < 8; ++r) {
      plds[wave][r + 8 * hi][m]      = (__bf16)p0[r];
      plds[wave][r + 8 * hi][m + 16] = (__bf16)p1[r];
    }
    __syncthreads();
    v16bf pa;
    #pragma unroll
    for (int v = 0; v < 8; ++v) {
      const int K = a_kidx(v, hi);
      pa[2 * v]     = plds[wave][m][K];
      pa[2 * v + 1] = plds[wave][m][K + 1];
    }

    // O += P @ V ; V B-operand from the TDM-staged LDS chunk (80B row stride)
    const unsigned char* vbb = &vstage[j & 1][0];
    #pragma unroll
    for (int t = 0; t < 16; ++t) {
      const v16bf vb = *(const v16bf*)(vbb + (t * 16 + m) * VROWB + hi * 32);
      acc[t] = __builtin_amdgcn_wmma_f32_16x16x32_bf16(false, pa, false, vb,
                                                       (short)0, acc[t], false, false);
    }
    __syncthreads();   // everyone done with buf (j&1) & plds before next overwrite
  }

  // epilogue: out = x + O / l
  #pragma unroll
  for (int r = 0; r < 8; ++r) {
    const float inv = 1.0f / lrun[r];
    const size_t rowbase = (size_t)(b * N_ + q0 + r + 8 * hi) * C_;
    #pragma unroll
    for (int t = 0; t < 16; ++t) {
      const size_t idx = rowbase + t * 16 + m;
      outp[idx] = x[idx] + acc[t][r] * inv;
    }
  }
}

// ---------------------------------------------------------------------------
extern "C" void kernel_launch(void* const* d_in, const int* in_sizes, int n_in,
                              void* d_out, int out_size, void* d_ws, size_t ws_size,
                              hipStream_t stream) {
  const float* x   = (const float*)d_in[0];
  const float* Wf  = (const float*)d_in[1];
  const float* bfp = (const float*)d_in[2];
  const float* Wg  = (const float*)d_in[3];
  const float* bgp = (const float*)d_in[4];
  const float* Wh  = (const float*)d_in[5];
  const float* bhp = (const float*)d_in[6];
  float* outp = (float*)d_out;

  __bf16* fbuf = (__bf16*)d_ws;                       // 1 MB
  __bf16* gbuf = fbuf + (size_t)B_ * N_ * CD_;        // 1 MB
  __bf16* hhT  = gbuf + (size_t)B_ * N_ * CD_;        // 8 MB

  proj_fg_kernel<<<dim3((B_ * N_) / 16, CD_ / 16, 2), 32, 0, stream>>>(
      x, Wf, bfp, Wg, bgp, fbuf, gbuf);
  proj_h_kernel<<<dim3((B_ * N_) / 16, C_ / 16), 32, 0, stream>>>(x, Wh, bhp, hhT);
  attn_kernel<<<dim3((B_ * N_) / 16 / 4), 128, 0, stream>>>(x, fbuf, gbuf, hhT, outp);
}
